// InfiniAttention_31593779429393
// MI455X (gfx1250) — matrix operand order repaired and checked by reading, not compiled, not run on hardware
//
#include <hip/hip_runtime.h>
#include <stdint.h>

// ---------------------------------------------------------------------------
// InfiniAttention for MI455X (gfx1250), wave32 + bf16 WMMA + TDM staging.
// Pipeline: f32->bf16 convert | QKV GEMMs (TDM double-buffered LDS staging) |
//           flash attention w/ relative bias + linear-attention memory read |
//           output projection GEMM.
// All matrix math on v_wmma_f32_16x16x32_bf16 (f32 accumulate).
// ---------------------------------------------------------------------------

#define DM 1024
#define NH 16
#define HD 64
#define BB 8
#define SS 1024
#define NROWS (BB * SS) // 8192

typedef __attribute__((ext_vector_type(16))) __bf16 v16bf;
typedef __attribute__((ext_vector_type(8)))  float  v8f;
typedef __attribute__((ext_vector_type(4)))  unsigned u32x4;
typedef __attribute__((ext_vector_type(8)))  int      i32x8;
typedef __attribute__((ext_vector_type(4)))  int      i32x4;

union Frag {
    v16bf    v;
    unsigned u[8];
};

__device__ __forceinline__ float bf2f(unsigned short h) {
    unsigned u = ((unsigned)h) << 16;
    return __builtin_bit_cast(float, u);
}
__device__ __forceinline__ unsigned short f2bf(float f) {
    unsigned u = __builtin_bit_cast(unsigned, f);
    u += 0x7fffu + ((u >> 16) & 1u); // round-to-nearest-even
    return (unsigned short)(u >> 16);
}
// A-matrix (16x32, 16-bit) per-lane K index for value-pair v (contiguous, matches B):
// lanes 0-15 hold K {0..15}, lanes 16-31 hold K {16..31}.
__device__ __forceinline__ int a_k(int v, int lh) {
    return (lh << 4) + (v << 1);
}
// Generic shared pointer -> LDS byte offset (flat LDS aperture uses addr[31:0]).
__device__ __forceinline__ unsigned lds_off(const void* p) {
    return (unsigned)(uintptr_t)p;
}

// ---------------------------------------------------------------------------
// Tensor Data Mover: 2D tile load global -> LDS (ISA ch.7-8, D# groups 0/1).
//   data_size = 2 bytes (bf16).  Tile laid out row-major (X fastest) in LDS,
//   matching the As/Bs layouts consumed by the WMMA fragment loads.
// ---------------------------------------------------------------------------
__device__ __forceinline__ void tdm_load_2d_bf16(
    unsigned lds_byte, const void* gtile,
    unsigned tensor_d0, unsigned tensor_d1,
    unsigned tile_d0, unsigned tile_d1,
    unsigned stride0_elems) {
    uint64_t ga = (uint64_t)(uintptr_t)gtile;
    u32x4 g0;
    g0[0] = 1u;                                  // count=1, user descriptor
    g0[1] = lds_byte;                            // lds_addr [63:32]
    g0[2] = (unsigned)ga;                        // global_addr [95:64]
    g0[3] = (unsigned)((ga >> 32) & 0x01FFFFFFu) // global_addr [120:96]
            | (2u << 30);                        // type=2 ("image") [127:126]
    i32x8 g1;
    g1[0] = (int)(1u << 16);                     // wg_mask=0, data_size=1 (2B)
    g1[1] = (int)((tensor_d0 & 0xFFFFu) << 16);  // tensor_dim0 lo
    g1[2] = (int)((tensor_d0 >> 16) | ((tensor_d1 & 0xFFFFu) << 16));
    g1[3] = (int)((tensor_d1 >> 16) | (tile_d0 << 16));
    g1[4] = (int)(tile_d1 & 0xFFFFu);            // tile_dim1; tile_dim2=0 (2D)
    g1[5] = (int)stride0_elems;                  // tensor_dim0_stride [31:0]
    g1[6] = 0;                                   // stride hi / dim1_stride lo
    g1[7] = 0;
    i32x4 z4 = {0, 0, 0, 0};
#if defined(__clang_major__) && (__clang_major__ >= 23)
    i32x8 z8 = {0, 0, 0, 0, 0, 0, 0, 0};
    __builtin_amdgcn_tensor_load_to_lds(g0, g1, z4, z4, z8, 0);
#else
    __builtin_amdgcn_tensor_load_to_lds(g0, g1, z4, z4, 0);
#endif
}

// ---------------------------------------------------------------------------
// Kernel 0: f32 -> bf16 conversion
// ---------------------------------------------------------------------------
__global__ void cvt_f32_bf16(const float* __restrict__ src,
                             unsigned short* __restrict__ dst, int n) {
    int i = blockIdx.x * 256 + threadIdx.x;
    if (i < n) dst[i] = f2bf(src[i]);
}

// ---------------------------------------------------------------------------
// Kernel 1: GEMM  C[n, o] = sum_k A[n, k] * W[o, k] + bias[o]
//   A: [8192, 1024] bf16 row-major, W: [1024, 1024] bf16 row-major (o-major,
//   i.e. torch Linear weight, so this computes x @ W.T).
//   mode 0: store bf16 at [(b*16+h)*1024 + s]*64 + d      (Q / K layout)
//   mode 1: store bf16 at [(b*16+h)*64 + d]*1024 + s      (V transposed)
//   mode 2: store f32  at [n*1024 + o]                    (final output)
// Block: 256 thr = 8 waves; block tile 128(M) x 64(N); K slabs of 32 staged
// into double-buffered LDS by the Tensor Data Mover (wave 0 issues, TENSORcnt
// tracks, s_wait_tensorcnt(2) overlaps next-slab DMA with current WMMAs).
// ---------------------------------------------------------------------------
__global__ __launch_bounds__(256) void gemm_bf16(
    const unsigned short* __restrict__ A,
    const unsigned short* __restrict__ W,
    const float* __restrict__ bias,
    void* __restrict__ dst, int mode) {
    __shared__ unsigned short As[2][128 * 32];
    __shared__ unsigned short Bs[2][64 * 32];

    const int t  = threadIdx.x;
    const int L  = t & 31;
    const int w  = t >> 5;      // wave id 0..7
    const int lh = L >> 4;      // lane half
    const int lm = L & 15;
    const int m0 = blockIdx.x * 128;
    const int n0 = blockIdx.y * 64;

    const unsigned aLds0 = lds_off(&As[0][0]), aLds1 = lds_off(&As[1][0]);
    const unsigned bLds0 = lds_off(&Bs[0][0]), bLds1 = lds_off(&Bs[1][0]);

    v8f acc[4] = {};

    if (w == 0) { // prime buffer 0 (TDM ignores EXEC; guard is wave-uniform)
        tdm_load_2d_bf16(aLds0, &A[(size_t)m0 * DM], DM, NROWS, 32, 128, DM);
        tdm_load_2d_bf16(bLds0, &W[(size_t)n0 * DM], DM, DM, 32, 64, DM);
    }

    int p = 0;
    for (int kk = 0; kk < DM; kk += 32) {
        if (w == 0) {
            if (kk + 32 < DM) { // prefetch next K slab into the other buffer
                tdm_load_2d_bf16(p ? aLds0 : aLds1,
                                 &A[(size_t)m0 * DM + kk + 32], DM, NROWS, 32, 128, DM);
                tdm_load_2d_bf16(p ? bLds0 : bLds1,
                                 &W[(size_t)n0 * DM + kk + 32], DM, DM, 32, 64, DM);
                __builtin_amdgcn_s_wait_tensorcnt(2); // current slab's 2 DMAs done
            } else {
                __builtin_amdgcn_s_wait_tensorcnt(0);
            }
        }
        __syncthreads(); // slab visible to all 8 waves

        const unsigned short* __restrict__ Ap = As[p];
        const unsigned short* __restrict__ Bp = Bs[p];
        Frag a;
#pragma unroll
        for (int v = 0; v < 8; ++v)
            a.u[v] = *(const unsigned*)&Ap[(w * 16 + lm) * 32 + a_k(v, lh)];

#pragma unroll
        for (int nt = 0; nt < 4; ++nt) {
            Frag b;
#pragma unroll
            for (int j = 0; j < 8; ++j)
                b.u[j] = *(const unsigned*)&Bp[(nt * 16 + lm) * 32 + (lh << 4) + (j << 1)];
            acc[nt] = __builtin_amdgcn_wmma_f32_16x16x32_bf16(
                false, a.v, false, b.v, (short)0, acc[nt], false, false);
        }
        __syncthreads(); // all reads done before this buffer is re-targeted
        p ^= 1;
    }

    // Epilogue.  C layout: lane L, reg r -> row r + (L/16)*8, col L%16.
#pragma unroll
    for (int nt = 0; nt < 4; ++nt) {
#pragma unroll
        for (int r = 0; r < 8; ++r) {
            int row = m0 + w * 16 + r + lh * 8; // n in [0, 8192)
            int col = n0 + nt * 16 + lm;        // o in [0, 1024)
            float val = acc[nt][r] + bias[col];
            if (mode == 2) {
                ((float*)dst)[(size_t)row * DM + col] = val;
            } else {
                int b = row >> 10, s = row & 1023, hh = col >> 6, d = col & 63;
                unsigned short bv = f2bf(val);
                if (mode == 0)
                    ((unsigned short*)dst)[((size_t)(b * NH + hh) * SS + s) * HD + d] = bv;
                else
                    ((unsigned short*)dst)[((size_t)(b * NH + hh) * HD + d) * SS + s] = bv;
            }
        }
    }
}

// ---------------------------------------------------------------------------
// Kernel 2: fused attention. One wave per (b,h, 16-query tile).
//   scores = (Q Kt + Q RelT) / 8 ; online softmax ; O = P V
//   + linear-attention read: sigma=elu(q)+1, num = sigma @ M[h], den = sigma.z
//   combined = gate*local + (1-gate)*num/den, stored bf16 as [b,s, h*64+d].
// Rel trick: dist = i-j+1024 never clips; per 32-key tile compute the 16x48
// window GEMM Cwin = Qt @ E[base..base+47]^T and redistribute with shuffles.
// ---------------------------------------------------------------------------
__global__ __launch_bounds__(32) void attn_kernel(
    const unsigned short* __restrict__ Qb,   // [B*H, S, 64]
    const unsigned short* __restrict__ Kb,   // [B*H, S, 64]
    const unsigned short* __restrict__ Vtb,  // [B*H, 64, S]
    const unsigned short* __restrict__ Eb,   // [2049, 64]
    const unsigned short* __restrict__ Mb,   // [16, 64, 64]
    const float* __restrict__ z,             // [16, 64]
    const float* __restrict__ galpha,        // [1]
    unsigned short* __restrict__ Yb) {       // [8192, 1024]
    __shared__ unsigned short Plds[16 * 32]; // probability tile staging

    const int L = threadIdx.x, lh = L >> 4, lm = L & 15;
    const int bh = blockIdx.y, h = bh & 15, b = bh >> 4;
    const int i0 = blockIdx.x * 16;

    // Q A-fragments (rows = queries, K = head dim, two 32-wide halves).
    Frag qa[2];
    const unsigned short* Qrow = Qb + ((size_t)bh * SS + i0 + lm) * HD;
#pragma unroll
    for (int f = 0; f < 2; ++f)
#pragma unroll
        for (int v = 0; v < 8; ++v)
            qa[f].u[v] = *(const unsigned*)&Qrow[a_k(v, lh) + f * 32];

    float mrow[8], lrow[8];
    v8f   o[4] = {};
#pragma unroll
    for (int r = 0; r < 8; ++r) { mrow[r] = -1e30f; lrow[r] = 0.f; }

    for (int j0 = 0; j0 < SS; j0 += 32) {
        // ---- QK^T: two 16x16 score tiles (keys j0..15, j0+16..31) ----
        v8f c0 = {}, c1 = {};
#pragma unroll
        for (int f = 0; f < 2; ++f) {
            Frag kb0, kb1;
            const unsigned short* Kr0 =
                Kb + ((size_t)bh * SS + j0 + lm) * HD + (lh << 4) + f * 32;
            const unsigned short* Kr1 = Kr0 + 16 * HD;
#pragma unroll
            for (int j = 0; j < 8; ++j) {
                kb0.u[j] = *(const unsigned*)&Kr0[j << 1];
                kb1.u[j] = *(const unsigned*)&Kr1[j << 1];
            }
            c0 = __builtin_amdgcn_wmma_f32_16x16x32_bf16(false, qa[f].v, false, kb0.v,
                                                         (short)0, c0, false, false);
            c1 = __builtin_amdgcn_wmma_f32_16x16x32_bf16(false, qa[f].v, false, kb1.v,
                                                         (short)0, c1, false, false);
        }

        // ---- relative-bias window: Cwin[di, m] = q[i0+di] . E[base2+m] ----
        const int base2 = i0 - j0 + 993; // i0-j0+1024-31, always in [1, 2001]
        v8f cw[3] = {};
#pragma unroll
        for (int wt = 0; wt < 3; ++wt)
#pragma unroll
            for (int f = 0; f < 2; ++f) {
                Frag eb;
                const unsigned short* Er =
                    Eb + ((size_t)(base2 + wt * 16 + lm)) * HD + (lh << 4) + f * 32;
#pragma unroll
                for (int j = 0; j < 8; ++j) eb.u[j] = *(const unsigned*)&Er[j << 1];
                cw[wt] = __builtin_amdgcn_wmma_f32_16x16x32_bf16(
                    false, qa[f].v, false, eb.v, (short)0, cw[wt], false, false);
            }

        // redistribute Cwin[di, di-dj+31] into score tiles, then scale by 1/8
#pragma unroll
        for (int r = 0; r < 8; ++r) {
            const int di = r + lh * 8;
            const float w0 = cw[0][r], w1 = cw[1][r], w2 = cw[2][r];
            { // tile 0: dj = lm
                int mm  = di - lm + 31;                 // [16, 46]
                int src = (mm & 15) | (lh << 4);        // stay within lane-half
                float a0 = __shfl(w0, src, 32);
                float a1 = __shfl(w1, src, 32);
                float a2 = __shfl(w2, src, 32);
                float rel = (mm < 16) ? a0 : ((mm < 32) ? a1 : a2);
                c0[r] = (c0[r] + rel) * 0.125f;
            }
            { // tile 1: dj = 16 + lm
                int mm  = di - lm + 15;                 // [0, 30]
                int src = (mm & 15) | (lh << 4);
                float a0 = __shfl(w0, src, 32);
                float a1 = __shfl(w1, src, 32);
                float rel = (mm < 16) ? a0 : a1;
                c1[r] = (c1[r] + rel) * 0.125f;
            }
        }

        // ---- online softmax over this 16x32 tile ----
#pragma unroll
        for (int r = 0; r < 8; ++r) {
            float tm = fmaxf(c0[r], c1[r]);
            tm = fmaxf(tm, __shfl_xor(tm, 1, 32));
            tm = fmaxf(tm, __shfl_xor(tm, 2, 32));
            tm = fmaxf(tm, __shfl_xor(tm, 4, 32));
            tm = fmaxf(tm, __shfl_xor(tm, 8, 32));
            float mnew = fmaxf(mrow[r], tm);
            float al   = __expf(mrow[r] - mnew);
            float p0   = __expf(c0[r] - mnew);
            float p1   = __expf(c1[r] - mnew);
            float rs   = p0 + p1;
            rs += __shfl_xor(rs, 1, 32);
            rs += __shfl_xor(rs, 2, 32);
            rs += __shfl_xor(rs, 4, 32);
            rs += __shfl_xor(rs, 8, 32);
            lrow[r] = lrow[r] * al + rs;
            mrow[r] = mnew;
#pragma unroll
            for (int nt = 0; nt < 4; ++nt) o[nt][r] = o[nt][r] * al;
            const int di = r + lh * 8;
            Plds[di * 32 + lm]      = f2bf(p0);
            Plds[di * 32 + 16 + lm] = f2bf(p1);
        }

        // ---- P V : A = P (16x32), B = V^T (contiguous per lane) ----
        Frag pa;
#pragma unroll
        for (int v = 0; v < 8; ++v)
            pa.u[v] = *(const unsigned*)&Plds[lm * 32 + a_k(v, lh)];
#pragma unroll
        for (int nt = 0; nt < 4; ++nt) {
            Frag vb;
            const unsigned short* Vr =
                Vtb + ((size_t)bh * HD + nt * 16 + lm) * SS + j0 + (lh << 4);
#pragma unroll
            for (int j = 0; j < 8; ++j) vb.u[j] = *(const unsigned*)&Vr[j << 1];
            o[nt] = __builtin_amdgcn_wmma_f32_16x16x32_bf16(
                false, pa.v, false, vb.v, (short)0, o[nt], false, false);
        }
    }

    // ---- linear-attention read: sigma = elu(q)+1 on the A fragments ----
    Frag sg[2];
    float dpart = 0.f;
#pragma unroll
    for (int f = 0; f < 2; ++f)
#pragma unroll
        for (int v = 0; v < 8; ++v) {
            int kk = a_k(v, lh) + f * 32;
            unsigned uq = qa[f].u[v];
            float x0 = bf2f((unsigned short)(uq & 0xffff));
            float x1 = bf2f((unsigned short)(uq >> 16));
            float s0 = (x0 > 0.f) ? (x0 + 1.f) : __expf(x0);
            float s1 = (x1 > 0.f) ? (x1 + 1.f) : __expf(x1);
            dpart += s0 * z[h * HD + kk] + s1 * z[h * HD + kk + 1];
            sg[f].u[v] = (unsigned)f2bf(s0) | ((unsigned)f2bf(s1) << 16);
        }
    float den = dpart + __shfl_xor(dpart, 16, 32) + 1e-6f; // per row m = lm

    v8f cn[4] = {};
#pragma unroll
    for (int nt = 0; nt < 4; ++nt)
#pragma unroll
        for (int f = 0; f < 2; ++f) {
            Frag mb;
#pragma unroll
            for (int j = 0; j < 8; ++j) {
                int kk = (lh << 4) + f * 32 + (j << 1);
                unsigned short a0 = Mb[h * 4096 + kk * HD + nt * 16 + lm];
                unsigned short a1 = Mb[h * 4096 + (kk + 1) * HD + nt * 16 + lm];
                mb.u[j] = (unsigned)a0 | ((unsigned)a1 << 16);
            }
            cn[nt] = __builtin_amdgcn_wmma_f32_16x16x32_bf16(
                false, sg[f].v, false, mb.v, (short)0, cn[nt], false, false);
        }

    // ---- gate, combine, store bf16 [b, s, h*64+d] ----
    const float ga   = galpha[0];
    const float gate = 1.f / (1.f + __expf(-ga));
#pragma unroll
    for (int r = 0; r < 8; ++r) {
        const int row = r + lh * 8;
        const float dc = __shfl(den, row, 32); // den lives at lane with lm == row
        const float li = lrow[r];
#pragma unroll
        for (int nt = 0; nt < 4; ++nt) {
            float local = o[nt][r] / li;
            float glob  = cn[nt][r] / dc;
            float out   = gate * local + (1.f - gate) * glob;
            Yb[((size_t)(b * SS + i0 + row)) * DM + h * HD + nt * 16 + lm] = f2bf(out);
        }
    }
}

// ---------------------------------------------------------------------------
// Host-side orchestration (workspace: ~93 MB of bf16 staging).
// ---------------------------------------------------------------------------
extern "C" void kernel_launch(void* const* d_in, const int* in_sizes, int n_in,
                              void* d_out, int out_size, void* d_ws, size_t ws_size,
                              hipStream_t stream) {
    (void)in_sizes; (void)n_in; (void)out_size; (void)ws_size;
    const float* x  = (const float*)d_in[0];
    const float* Wq = (const float*)d_in[1];
    const float* bq = (const float*)d_in[2];
    const float* Wk = (const float*)d_in[3];
    const float* bk = (const float*)d_in[4];
    const float* Wv = (const float*)d_in[5];
    const float* bv = (const float*)d_in[6];
    const float* Wo = (const float*)d_in[7];
    const float* bo = (const float*)d_in[8];
    const float* E  = (const float*)d_in[9];
    const float* M  = (const float*)d_in[10];
    const float* z  = (const float*)d_in[11];
    const float* ga = (const float*)d_in[12];

    char*  ws  = (char*)d_ws;
    size_t off = 0;
    auto alloc = [&](size_t bytes) -> unsigned short* {
        off = (off + 255) & ~(size_t)255;
        char* p = ws + off;
        off += bytes;
        return (unsigned short*)p;
    };
    unsigned short* Xb  = alloc((size_t)NROWS * DM * 2);
    unsigned short* Wqb = alloc((size_t)DM * DM * 2);
    unsigned short* Wkb = alloc((size_t)DM * DM * 2);
    unsigned short* Wvb = alloc((size_t)DM * DM * 2);
    unsigned short* Wob = alloc((size_t)DM * DM * 2);
    unsigned short* Qb  = alloc((size_t)BB * NH * SS * HD * 2);
    unsigned short* Kbf = alloc((size_t)BB * NH * SS * HD * 2);
    unsigned short* Vtb = alloc((size_t)BB * NH * SS * HD * 2);
    unsigned short* Yb  = alloc((size_t)NROWS * DM * 2);
    unsigned short* Eb  = alloc((size_t)2049 * HD * 2);
    unsigned short* Mb  = alloc((size_t)NH * HD * HD * 2);

    auto cvt = [&](const float* s, unsigned short* d, int n) {
        cvt_f32_bf16<<<dim3((n + 255) / 256), dim3(256), 0, stream>>>(s, d, n);
    };
    cvt(x,  Xb,  NROWS * DM);
    cvt(Wq, Wqb, DM * DM);
    cvt(Wk, Wkb, DM * DM);
    cvt(Wv, Wvb, DM * DM);
    cvt(Wo, Wob, DM * DM);
    cvt(E,  Eb,  2049 * HD);
    cvt(M,  Mb,  NH * HD * HD);

    dim3 gg(NROWS / 128, DM / 64);
    gemm_bf16<<<gg, dim3(256), 0, stream>>>(Xb, Wqb, bq, (void*)Qb,  0);
    gemm_bf16<<<gg, dim3(256), 0, stream>>>(Xb, Wkb, bk, (void*)Kbf, 0);
    gemm_bf16<<<gg, dim3(256), 0, stream>>>(Xb, Wvb, bv, (void*)Vtb, 1);

    attn_kernel<<<dim3(SS / 16, BB * NH), dim3(32), 0, stream>>>(
        Qb, Kbf, Vtb, Eb, Mb, z, ga, Yb);

    gemm_bf16<<<gg, dim3(256), 0, stream>>>(Yb, Wob, bo, d_out, 2);
}
